// causal_attention_34548716929700
// MI455X (gfx1250) — matrix operand order repaired and checked by reading, not compiled, not run on hardware
//
#include <hip/hip_runtime.h>
#include <hip/hip_bf16.h>

// ---- CDNA5 WMMA types ----
typedef __attribute__((ext_vector_type(16))) __bf16 v16bf;
typedef __attribute__((ext_vector_type(8)))  __bf16 v8bf;
typedef __attribute__((ext_vector_type(8)))  float  v8f;

static __device__ __forceinline__ v8f wmma_bf16(v16bf a, v16bf b, v8f c) {
    // (neg_a, A, neg_b, B, c_mod, C, reuse_a, reuse_b)
    return __builtin_amdgcn_wmma_f32_16x16x32_bf16(false, a, false, b, (short)0, c, false, false);
}

static __device__ __forceinline__ float fast_exp2(float x) {
#if __has_builtin(__builtin_amdgcn_exp2f)
    return __builtin_amdgcn_exp2f(x);   // bare v_exp_f32
#else
    return exp2f(x);
#endif
}

static __device__ __forceinline__ v16bf cat8(v8bf lo, v8bf hi) {
    v16bf r;
#pragma unroll
    for (int i = 0; i < 8; ++i) { r[i] = lo[i]; r[8 + i] = hi[i]; }
    return r;
}

// Load 16 contiguous bf16 (32B) as one v16bf (element order == K order)
static __device__ __forceinline__ v16bf ldB16(const __bf16* p) {
    return cat8(*(const v8bf*)p, *(const v8bf*)(p + 8));
}

#define B_      4
#define S_      4096
#define DIN_    768
#define DOUT_   128
#define KT_     (DIN_ / 32)     // 24 k-tiles for the projection
#define NT_     (DOUT_ / 16)    // 8 n-tiles

// ------------------------------------------------------------------
// Kernel 1: pack W[768,128] (f32, row-major) into WMMA B-operand bf16
// layout: flat[((m*24+kt)*8+nt)*512 + lane*16 + j]
//   lane<16 : col n = nt*16+lane,    K = kt*32 + j        (j=0..15)
//   lane>=16: col n = nt*16+lane-16, K = kt*32 + 16 + j
// ------------------------------------------------------------------
__global__ void pack_w_kernel(const float* __restrict__ Wq,
                              const float* __restrict__ Wk,
                              const float* __restrict__ Wv,
                              __bf16* __restrict__ packed) {
    int tid = blockIdx.x * 256 + threadIdx.x;
    if (tid >= 3 * KT_ * NT_ * 32) return;
    int m    = tid / (KT_ * NT_ * 32);
    int r    = tid % (KT_ * NT_ * 32);
    int kt   = r / (NT_ * 32);
    int r2   = r % (NT_ * 32);
    int nt   = r2 / 32;
    int lane = r2 % 32;
    const float* W = (m == 0) ? Wq : (m == 1) ? Wk : Wv;
    int n     = nt * 16 + (lane & 15);
    int kBase = kt * 32 + ((lane < 16) ? 0 : 16);
    __bf16* dst = packed + (size_t)tid * 16;
#pragma unroll
    for (int j = 0; j < 16; ++j)
        dst[j] = (__bf16)W[(size_t)(kBase + j) * DOUT_ + n];
}

// ------------------------------------------------------------------
// Kernel 2: QKV projection.  grid = (16384/16, 3), block = 128 (4 waves)
// Each wave: 16 rows x 32 cols tile, K-loop of 24 WMMAs.
// ------------------------------------------------------------------
__global__ __launch_bounds__(128, 1)
void qkv_kernel(const float* __restrict__ x, const __bf16* __restrict__ packedW,
                __bf16* __restrict__ q, __bf16* __restrict__ k, __bf16* __restrict__ v) {
    const int lane = threadIdx.x & 31;
    const int wave = threadIdx.x >> 5;          // 0..3 -> n-tiles 2w, 2w+1
    const int mrow = lane & 15;
    const bool hi  = lane >= 16;
    const int acb  = hi ? 16 : 0;               // A-operand K chunk base
    const int row0 = blockIdx.x * 16;           // over B*S = 16384 rows
    const int mat  = blockIdx.y;                // 0=q 1=k 2=v

    const float*  xrow  = x + (size_t)(row0 + mrow) * DIN_;
    const __bf16* wbase = packedW + (size_t)mat * KT_ * NT_ * 512;

    v8f acc0 = {}; v8f acc1 = {};
#pragma unroll 4
    for (int kt = 0; kt < KT_; ++kt) {
        // A operand: lane<16 -> K {c..c+15} with c = kt*32
        //            lane>=16 -> shifted by 16
        const int ko = kt * 32 + acb;
        float4 a0 = *(const float4*)(xrow + ko);
        float4 a1 = *(const float4*)(xrow + ko + 4);
        float4 a2 = *(const float4*)(xrow + ko + 8);
        float4 a3 = *(const float4*)(xrow + ko + 12);
        if (kt + 1 < KT_) __builtin_prefetch(xrow + ko + 32, 0, 3);  // global_prefetch_b8
        v16bf A;
        A[0]=(__bf16)a0.x; A[1]=(__bf16)a0.y; A[2]=(__bf16)a0.z; A[3]=(__bf16)a0.w;
        A[4]=(__bf16)a1.x; A[5]=(__bf16)a1.y; A[6]=(__bf16)a1.z; A[7]=(__bf16)a1.w;
        A[8]=(__bf16)a2.x; A[9]=(__bf16)a2.y; A[10]=(__bf16)a2.z; A[11]=(__bf16)a2.w;
        A[12]=(__bf16)a3.x; A[13]=(__bf16)a3.y; A[14]=(__bf16)a3.z; A[15]=(__bf16)a3.w;

        const __bf16* bp0 = wbase + ((size_t)(kt * NT_ + wave * 2 + 0)) * 512 + lane * 16;
        const __bf16* bp1 = wbase + ((size_t)(kt * NT_ + wave * 2 + 1)) * 512 + lane * 16;
        acc0 = wmma_bf16(A, ldB16(bp0), acc0);
        acc1 = wmma_bf16(A, ldB16(bp1), acc1);
    }

    __bf16* dst = (mat == 0) ? q : (mat == 1) ? k : v;
    const int h8 = hi ? 8 : 0;
#pragma unroll
    for (int i = 0; i < 8; ++i) {
        size_t r = (size_t)(row0 + i + h8) * DOUT_;
        dst[r + (wave * 2 + 0) * 16 + mrow] = (__bf16)acc0[i];
        dst[r + (wave * 2 + 1) * 16 + mrow] = (__bf16)acc1[i];
    }
}

// ------------------------------------------------------------------
// Kernel 3: V[b,s,128] -> Vt[b,128,s]  (bf16, LDS tile transpose)
// grid = (S/64, 128/64, B), block = 256
// ------------------------------------------------------------------
__global__ void transpose_v_kernel(const __bf16* __restrict__ v, __bf16* __restrict__ vt) {
    __shared__ __bf16 t[64 * 65];
    const int b  = blockIdx.z;
    const int s0 = blockIdx.x * 64, d0 = blockIdx.y * 64;
    const __bf16* vb  = v  + (size_t)b * S_ * DOUT_;
    __bf16*       vtb = vt + (size_t)b * DOUT_ * S_;
    for (int idx = threadIdx.x; idx < 64 * 64; idx += 256) {
        int sl = idx >> 6, dl = idx & 63;
        t[dl * 65 + sl] = vb[(size_t)(s0 + sl) * DOUT_ + d0 + dl];
    }
    __syncthreads();
    for (int idx = threadIdx.x; idx < 64 * 64; idx += 256) {
        int dl = idx >> 6, sl = idx & 63;
        vtb[(size_t)(d0 + dl) * S_ + s0 + sl] = t[dl * 65 + sl];
    }
}

// ------------------------------------------------------------------
// Kernel 4: flash attention. grid = (S/64, B), block = 128 (4 waves).
// Each wave: 16 query rows, streams 32-key blocks with online softmax.
// Blocks are launched high-q first: long causal tiles start early and
// short tiles backfill, shrinking the makespan tail.
// ------------------------------------------------------------------
__global__ __launch_bounds__(128, 1)
void attn_kernel(const __bf16* __restrict__ q, const __bf16* __restrict__ kmat,
                 const __bf16* __restrict__ vt, float* __restrict__ out) {
    __shared__ float pbuf[4][16 * 32];          // per-wave P bounce buffer

    const int  lane = threadIdx.x & 31;
    const int  wave = threadIdx.x >> 5;
    const int  mrow = lane & 15;
    const bool hi   = lane >= 16;
    const int  acb  = hi ? 16 : 0;
    const int  h8   = hi ? 8 : 0;
    const int  b    = blockIdx.y;
    const int  qblk = (int)gridDim.x - 1 - (int)blockIdx.x;   // reversed order
    const int  qw   = qblk * 64 + wave * 16;    // first query row of this wave

    const __bf16* qb  = q    + (size_t)b * S_ * DOUT_;
    const __bf16* kb_ = kmat + (size_t)b * S_ * DOUT_;
    const __bf16* vtb = vt   + (size_t)b * DOUT_ * S_;

    // Q A-operand tiles (d split into 4 chunks of 32), resident in VGPRs
    v16bf Qa[4];
    const __bf16* qrow = qb + (size_t)(qw + mrow) * DOUT_;
#pragma unroll
    for (int dt = 0; dt < 4; ++dt) {
        const __bf16* p = qrow + dt * 32 + acb;
        Qa[dt] = cat8(*(const v8bf*)p, *(const v8bf*)(p + 8));
    }

    v8f   O[NT_];
    float mstat[8], lstat[8];
#pragma unroll
    for (int nt = 0; nt < NT_; ++nt) O[nt] = (v8f){};
#pragma unroll
    for (int i = 0; i < 8; ++i) { mstat[i] = -3.0e38f; lstat[i] = 0.0f; }

    const float SC = 0.12751745f;               // log2(e) / sqrt(128)
    const int   kbLast = (qw + 15) >> 5;
    float* pl = pbuf[wave];

    for (int kb = 0; kb <= kbLast; ++kb) {
        const int k0 = kb * 32;

        // prefetch next key block's K rows (global_prefetch_b8)
        if (kb < kbLast) {
            __builtin_prefetch(kb_ + (size_t)(k0 + 32 + mrow) * DOUT_, 0, 3);
            __builtin_prefetch(kb_ + (size_t)(k0 + 48 + mrow) * DOUT_, 0, 3);
        }

        // ---- scores S = Q K^T : two 16-key n-tiles, 4 d-chunks each ----
        v8f s0 = (v8f){}, s1 = (v8f){};
#pragma unroll
        for (int dt = 0; dt < 4; ++dt) {
            const __bf16* p0 = kb_ + (size_t)(k0 + mrow)      * DOUT_ + dt * 32 + (hi ? 16 : 0);
            const __bf16* p1 = kb_ + (size_t)(k0 + 16 + mrow) * DOUT_ + dt * 32 + (hi ? 16 : 0);
            s0 = wmma_bf16(Qa[dt], ldB16(p0), s0);
            s1 = wmma_bf16(Qa[dt], ldB16(p1), s1);
        }
#pragma unroll
        for (int i = 0; i < 8; ++i) { s0[i] *= SC; s1[i] *= SC; }

        // causal mask: only the diagonal block needs it
        if (kb == kbLast) {
#pragma unroll
            for (int i = 0; i < 8; ++i) {
                int qi = qw + i + h8;
                s0[i] = (k0 + mrow      > qi) ? -1.0e30f : s0[i];
                s1[i] = (k0 + 16 + mrow > qi) ? -1.0e30f : s1[i];
            }
        }

        // ---- online softmax (base 2), row reductions across 16-lane halves ----
        float rm[8];
#pragma unroll
        for (int i = 0; i < 8; ++i) rm[i] = fmaxf(s0[i], s1[i]);
#pragma unroll
        for (int off = 8; off >= 1; off >>= 1)
#pragma unroll
            for (int i = 0; i < 8; ++i)
                rm[i] = fmaxf(rm[i], __shfl_xor(rm[i], off, 32));

        float al[8];
#pragma unroll
        for (int i = 0; i < 8; ++i) {
            float mn = fmaxf(mstat[i], rm[i]);
            al[i] = fast_exp2(mstat[i] - mn);
            mstat[i] = mn;
        }
        float rs[8];
#pragma unroll
        for (int i = 0; i < 8; ++i) {
            s0[i] = fast_exp2(s0[i] - mstat[i]);
            s1[i] = fast_exp2(s1[i] - mstat[i]);
            rs[i] = s0[i] + s1[i];
        }
#pragma unroll
        for (int off = 8; off >= 1; off >>= 1)
#pragma unroll
            for (int i = 0; i < 8; ++i)
                rs[i] += __shfl_xor(rs[i], off, 32);
#pragma unroll
        for (int i = 0; i < 8; ++i) lstat[i] = lstat[i] * al[i] + rs[i];
#pragma unroll
        for (int nt = 0; nt < NT_; ++nt)
#pragma unroll
            for (int i = 0; i < 8; ++i) O[nt][i] *= al[i];

        // ---- relayout P: C layout -> A operand, through per-wave LDS ----
#pragma unroll
        for (int i = 0; i < 8; ++i) {
            pl[(i + h8) * 32 + mrow]      = s0[i];
            pl[(i + h8) * 32 + 16 + mrow] = s1[i];
        }
        const float* ap = pl + mrow * 32 + acb;
        float4 c0 = *(const float4*)(ap);
        float4 c1 = *(const float4*)(ap + 4);
        float4 c2 = *(const float4*)(ap + 8);
        float4 c3 = *(const float4*)(ap + 12);
        v16bf Pa;
        Pa[0]=(__bf16)c0.x; Pa[1]=(__bf16)c0.y; Pa[2]=(__bf16)c0.z; Pa[3]=(__bf16)c0.w;
        Pa[4]=(__bf16)c1.x; Pa[5]=(__bf16)c1.y; Pa[6]=(__bf16)c1.z; Pa[7]=(__bf16)c1.w;
        Pa[8]=(__bf16)c2.x; Pa[9]=(__bf16)c2.y; Pa[10]=(__bf16)c2.z; Pa[11]=(__bf16)c2.w;
        Pa[12]=(__bf16)c3.x; Pa[13]=(__bf16)c3.y; Pa[14]=(__bf16)c3.z; Pa[15]=(__bf16)c3.w;

        // ---- O += P @ V : Vt[d, s] rows are contiguous in key => clean B loads ----
#pragma unroll
        for (int nt = 0; nt < NT_; ++nt) {
            const __bf16* vp = vtb + (size_t)(nt * 16 + mrow) * S_ + k0 + (hi ? 16 : 0);
            O[nt] = wmma_bf16(Pa, ldB16(vp), O[nt]);
        }
    }

    // ---- epilogue: divide by row sums, store f32 ----
    float inv[8];
#pragma unroll
    for (int i = 0; i < 8; ++i) inv[i] = 1.0f / lstat[i];
    const size_t obase = (size_t)b * S_ * DOUT_;
#pragma unroll
    for (int nt = 0; nt < NT_; ++nt)
#pragma unroll
        for (int i = 0; i < 8; ++i)
            out[obase + (size_t)(qw + i + h8) * DOUT_ + nt * 16 + mrow] = O[nt][i] * inv[i];
}

// ------------------------------------------------------------------
extern "C" void kernel_launch(void* const* d_in, const int* in_sizes, int n_in,
                              void* d_out, int out_size, void* d_ws, size_t ws_size,
                              hipStream_t stream) {
    (void)in_sizes; (void)n_in; (void)out_size; (void)ws_size;
    const float* x  = (const float*)d_in[0];
    const float* Wq = (const float*)d_in[1];
    const float* Wk = (const float*)d_in[2];
    const float* Wv = (const float*)d_in[3];
    float* out = (float*)d_out;

    // workspace layout (bytes): packedW | q | k | v | vt
    char* ws = (char*)d_ws;
    const size_t PACKW_B = (size_t)3 * KT_ * NT_ * 512 * sizeof(__bf16);   // 589,824
    const size_t MAT_E   = (size_t)B_ * S_ * DOUT_;                        // 2,097,152 elems
    __bf16* packedW = (__bf16*)ws;
    __bf16* q  = (__bf16*)(ws + PACKW_B);
    __bf16* k  = q + MAT_E;
    __bf16* v  = k + MAT_E;
    __bf16* vt = v + MAT_E;

    pack_w_kernel<<<(3 * KT_ * NT_ * 32 + 255) / 256, 256, 0, stream>>>(Wq, Wk, Wv, packedW);
    qkv_kernel<<<dim3((B_ * S_) / 16, 3), 128, 0, stream>>>(x, packedW, q, k, v);
    transpose_v_kernel<<<dim3(S_ / 64, DOUT_ / 64, B_), 256, 0, stream>>>(v, vt);
    attn_kernel<<<dim3(S_ / 64, B_), 128, 0, stream>>>(q, k, vt, out);
}
